// GNNLayer_23716809408745
// MI455X (gfx1250) — matrix operand order repaired and checked by reading, not compiled, not run on hardware
//
#include <hip/hip_runtime.h>
#include <hip/hip_bf16.h>

typedef __attribute__((ext_vector_type(16))) _Float16 v16h;
typedef __attribute__((ext_vector_type(8)))  _Float16 v8h;
typedef __attribute__((ext_vector_type(8)))  float    v8f;

#define D_    64
#define MEM_  4
#define NREL_ 8
#define EPW   64     // edges (or nodes) per workgroup
#define BLK   128    // 4 wave32
#define LDX   72     // f16 leading dim (mult of 8 halves -> 16B aligned, conflict-free)
#define LDKV  136    // f16 leading dim for K|V buffer
#define LDF   40     // f16 leading dim for FF buffer (K padded 16->32)
#define LDY   66     // f32 leading dim (scalar access only)

// ---------------- WMMA fragment helpers (CDNA5 16x16x32 f16) ----------------

// A fragment: lane l holds row m=l&15, K = {kb..kb+7, kb+16..kb+23}, kb=(l>>4)*8
__device__ __forceinline__ v16h fragA_ld(const _Float16* p, int ld, unsigned l) {
  const unsigned m = l & 15u, kb = (l >> 4) * 8u;
  const _Float16* row = p + (size_t)m * ld + kb;
  v8h lo = *(const v8h*)row;
  v8h hi = *(const v8h*)(row + 16);
  v16h f;
#pragma unroll
  for (int i = 0; i < 8; ++i) { f[i] = lo[i]; f[8 + i] = hi[i]; }
  return f;
}

// B fragment (from row-major W[N][K]): lane l holds col n=l&15, K = {kb..kb+7, kb+16..kb+23}, kb=(l>>4)*8
__device__ __forceinline__ v16h fragB_ld(const _Float16* p, int ld, unsigned l) {
  const unsigned n = l & 15u, kb = (l >> 4) * 8u;
  const _Float16* row = p + (size_t)n * ld + kb;
  v8h lo = *(const v8h*)row;
  v8h hi = *(const v8h*)(row + 16);
  v16h f;
#pragma unroll
  for (int i = 0; i < 8; ++i) { f[i] = lo[i]; f[8 + i] = hi[i]; }
  return f;
}

__device__ __forceinline__ v8f wmma16(v16h a, v16h b, v8f c) {
  return __builtin_amdgcn_wmma_f32_16x16x32_f16(false, a, false, b, (short)0, c,
                                                false, false);
}

__device__ __forceinline__ v16h scale16(v16h a, _Float16 s) {
  v16h r;
#pragma unroll
  for (int i = 0; i < 16; ++i) r[i] = a[i] * s;  // -> v_pk_mul_f16
  return r;
}

// ---------------- small prep kernels ----------------

__global__ void prep_weights(const float* __restrict__ Wqkv, const float* __restrict__ Wo,
                             const float* __restrict__ W1, const float* __restrict__ W2,
                             const float* __restrict__ Wc, const float* __restrict__ CLS,
                             const float* __restrict__ SEP, const float* __restrict__ Wm,
                             const float* __restrict__ feat,
                             _Float16* hWqkv, _Float16* hWo, _Float16* hW1, _Float16* hW2p,
                             _Float16* hWcP, _Float16* hCLS, _Float16* hSEP,
                             _Float16* WmR, _Float16* featH, int nN) {
  int gid = blockIdx.x * blockDim.x + threadIdx.x;
  int stride = gridDim.x * blockDim.x;
  for (int i = gid; i < nN * D_; i += stride) featH[i] = (_Float16)feat[i];
  // WmR[j][o][i] = Wm[o*64+i][j]   (B-operand of the hypernetwork GEMM)
  for (int i = gid; i < 64 * 64 * 64; i += stride) {
    int j = i >> 12, o = (i >> 6) & 63, ii = i & 63;
    WmR[i] = (_Float16)Wm[(size_t)(o * 64 + ii) * 64 + j];
  }
  for (int i = gid; i < 192 * 64; i += stride) hWqkv[i] = (_Float16)Wqkv[i];
  for (int i = gid; i < 64 * 64; i += stride) hWo[i] = (_Float16)Wo[i];
  for (int i = gid; i < 16 * 64; i += stride) hW1[i] = (_Float16)W1[i];
  for (int i = gid; i < 64 * 32; i += stride) {  // W2 [64][16] padded K->32 with zeros
    int o = i >> 5, k = i & 31;
    hW2p[i] = (_Float16)(k < 16 ? W2[o * 16 + k] : 0.f);
  }
  for (int i = gid; i < 16 * 64; i += stride) {  // Wc [4][64] padded N->16 with zeros
    int o = i >> 6, k = i & 63;
    hWcP[i] = (_Float16)(o < MEM_ ? Wc[o * 64 + k] : 0.f);
  }
  for (int i = gid; i < MEM_ * D_; i += stride) hCLS[i] = (_Float16)CLS[i];
  for (int i = gid; i < D_; i += stride) hSEP[i] = (_Float16)SEP[i];
}

__global__ void zero_ws(float* agg, float* cnt, int nN) {
  int gid = blockIdx.x * blockDim.x + threadIdx.x;
  int stride = gridDim.x * blockDim.x;
  for (int i = gid; i < nN * D_; i += stride) agg[i] = 0.f;
  for (int i = gid; i < nN; i += stride) cnt[i] = 0.f;
}

__global__ void count_edges(const int* __restrict__ dst, float* cnt, int nE) {
  int e = blockIdx.x * blockDim.x + threadIdx.x;
  if (e < nE) atomicAdd(&cnt[dst[e]], 1.f);
}

// stable per-etype rank via wave32 ballot scan (single block)
__global__ void rank_kernel(const int* __restrict__ etype, int nE,
                            int* rank, int* counts, int* start) {
  __shared__ int base[NREL_];
  __shared__ int waveCnt[8][NREL_];
  int tid = threadIdx.x;
  int l = tid & 31, wv = tid >> 5;
  if (tid < NREL_) base[tid] = 0;
  __syncthreads();
  for (int c0 = 0; c0 < nE; c0 += 256) {
    int e = c0 + tid;
    int et = (e < nE) ? etype[e] : -1;
    int myrank = 0;
    for (int t = 0; t < NREL_; ++t) {
      unsigned long long mask = __ballot(et == t);
      unsigned m32 = (unsigned)mask;
      if (et == t) myrank = __popc(m32 & ((1u << l) - 1u));
      if (l == 0) waveCnt[wv][t] = __popc(m32);
    }
    __syncthreads();
    if (et >= 0) {
      int off = base[et];
      for (int w2 = 0; w2 < wv; ++w2) off += waveCnt[w2][et];
      rank[e] = off + myrank;
    }
    __syncthreads();
    if (tid < NREL_) {
      int tot = 0;
      for (int w2 = 0; w2 < 8; ++w2) tot += waveCnt[w2][tid];
      base[tid] += tot;
    }
    __syncthreads();
  }
  if (tid < NREL_) counts[tid] = base[tid];
  __syncthreads();
  if (tid == 0) {
    int acc = 0;
    for (int t = 0; t < NREL_; ++t) { start[t] = acc; acc += base[t]; }
  }
}

// ---------------- main fused kernel ----------------

struct GP {
  const _Float16 *featH, *WmR, *hWqkv, *hWo, *hW1, *hW2p, *hWcP, *hCLS, *hSEP;
  const float *bqkv, *bo, *b1, *b2, *ln1w, *ln1b, *ln2w, *ln2b, *bc, *h_bias;
  const int *src_idx, *dst_idx, *etype, *rank, *counts, *start;
  float *agg, *cnt;
  float *node_rep;
  int *label;
  float *category;
  int nE, nN;
};

template <bool SELF>
__global__ __launch_bounds__(BLK) void gnn_main(GP p) {
  __shared__ _Float16 sSrc[EPW][LDX];   // src rows for all 64 edges (also hypernet A)
  __shared__ _Float16 sCLS[16][LDX];    // token0 rows (per mem pass)
  __shared__ _Float16 sSEP[16][LDX];    // token2 rows (replicated SEP)
  __shared__ _Float16 sDst[16][LDX];    // token3 rows (per sub-tile)
  __shared__ _Float16 sQ[16][LDX];      // Q of CLS rows
  __shared__ _Float16 sKV[64][LDKV];    // K (cols 0..63) | V (cols 64..127), all 64 rows
  __shared__ _Float16 sAT[16][LDX];     // attention out / LN2 tmp (disjoint lifetimes)
  __shared__ float    sY[16][LDY];      // pre-LN f32 rows
  __shared__ _Float16 sH[16][LDX];      // post-LN1 hidden
  __shared__ _Float16 sF[16][LDF];      // FF1 out, K padded to 32
  __shared__ float    sMem[EPW][LDY];   // mem accumulator (f32) for all 64 edges

  const int tid = threadIdx.x;
  const unsigned l = (unsigned)(tid & 31);
  const int wv = tid >> 5;
  const int base = blockIdx.x * EPW;
  const int total = SELF ? p.nN : p.nE;

  // ---- S0: gather src rows, SEP rows, zero mem accumulator + FF pad cols ----
  for (int idx = tid; idx < EPW * 8; idx += BLK) {
    int el = idx >> 3, cc = (idx & 7) * 8;
    int e = base + el;
    int ec = e < total ? e : total - 1;
    int sr = SELF ? ec : p.src_idx[ec];
    *(v8h*)&sSrc[el][cc] = *(const v8h*)(p.featH + (size_t)sr * D_ + cc);
  }
  for (int idx = tid; idx < 16 * 8; idx += BLK) {
    int s = idx >> 3, cc = (idx & 7) * 8;
    *(v8h*)&sSEP[s][cc] = *(const v8h*)(p.hSEP + cc);
  }
  for (int idx = tid; idx < EPW * D_; idx += BLK) sMem[idx >> 6][idx & 63] = 0.f;
  for (int idx = tid; idx < 16 * 16; idx += BLK)  // FF1 only writes cols 0..15
    sF[idx >> 4][16 + (idx & 15)] = (_Float16)0.f;
  __syncthreads();

  for (int st = 0; st < 4; ++st) {
    // ---- token3 (dst) rows for this sub-tile ----
    for (int idx = tid; idx < 16 * 8; idx += BLK) {
      int s = idx >> 3, cc = (idx & 7) * 8;
      int e = base + st * 16 + s;
      int ec = e < total ? e : total - 1;
      int dr = SELF ? ec : p.dst_idx[ec];
      *(v8h*)&sDst[s][cc] = *(const v8h*)(p.featH + (size_t)dr * D_ + cc);
    }
    __syncthreads();

    // ---- S2: K,V for token rows 1..3, computed ONCE per sub-tile (24 tiles) ----
    for (int t6 = 0; t6 < 6; ++t6) {
      int tile = wv * 6 + t6;
      int mt = 1 + (tile % 3);        // token index
      int nt = 4 + (tile / 3);        // QKV column tile (K,V region)
      const _Float16* abase =
          (mt == 1) ? &sSrc[st * 16][0] : (mt == 2) ? &sSEP[0][0] : &sDst[0][0];
      v8f acc = {};
#pragma unroll
      for (int kk = 0; kk < 2; ++kk) {
        v16h a = fragA_ld(abase + kk * 32, LDX, l);
        v16h b = fragB_ld(p.hWqkv + (size_t)(nt * 16) * D_ + kk * 32, D_, l);
        acc = wmma16(a, b, acc);
      }
      int n = nt * 16 + (int)(l & 15u);
      float bb = p.bqkv[n];
      int mb = mt * 16 + (int)((l >> 4) * 8u);
#pragma unroll
      for (int r = 0; r < 8; ++r) sKV[mb + r][n - 64] = (_Float16)(acc[r] + bb);
    }

    for (int mi = 0; mi < MEM_; ++mi) {
      // ---- CLS rows for this pass ----
      for (int idx = tid; idx < 16 * 8; idx += BLK) {
        int s = idx >> 3, cc = (idx & 7) * 8;
        *(v8h*)&sCLS[s][cc] = *(const v8h*)(p.hCLS + mi * D_ + cc);
      }
      __syncthreads();

      // ---- S3: full QKV for CLS rows (12 tiles, 3 per wave) ----
      for (int t3 = 0; t3 < 3; ++t3) {
        int nt = wv * 3 + t3;
        v8f acc = {};
#pragma unroll
        for (int kk = 0; kk < 2; ++kk) {
          v16h a = fragA_ld(&sCLS[0][kk * 32], LDX, l);
          v16h b = fragB_ld(p.hWqkv + (size_t)(nt * 16) * D_ + kk * 32, D_, l);
          acc = wmma16(a, b, acc);
        }
        int n = nt * 16 + (int)(l & 15u);
        float bb = p.bqkv[n];
        int mb = (int)((l >> 4) * 8u);
#pragma unroll
        for (int r = 0; r < 8; ++r) {
          float v = acc[r] + bb;
          if (nt < 4) sQ[mb + r][n] = (_Float16)v;
          else        sKV[mb + r][n - 64] = (_Float16)v;
        }
      }
      __syncthreads();

      // ---- S4: attention (CLS query only); lane pair per (seq, head) unit ----
      {
        int u = tid >> 1, half = tid & 1;
        int s = u & 15, h = u >> 4;
        int d0 = half * 8;
        float q[8];
#pragma unroll
        for (int d = 0; d < 8; ++d) q[d] = (float)sQ[s][h * 16 + d0 + d];
        float sc[4];
#pragma unroll
        for (int kt = 0; kt < 4; ++kt) {
          float a = 0.f;
#pragma unroll
          for (int d = 0; d < 8; ++d)
            a += q[d] * (float)sKV[kt * 16 + s][h * 16 + d0 + d];
          a += __shfl_xor(a, 1, 32);  // combine the two 8-dim halves
          sc[kt] = a * 0.25f;         // 1/sqrt(16)
        }
        float mx = fmaxf(fmaxf(sc[0], sc[1]), fmaxf(sc[2], sc[3]));
        float pr[4], sum = 0.f;
#pragma unroll
        for (int kt = 0; kt < 4; ++kt) { pr[kt] = __expf(sc[kt] - mx); sum += pr[kt]; }
        float inv = 1.f / sum;
#pragma unroll
        for (int d = 0; d < 8; ++d) {
          float o = 0.f;
#pragma unroll
          for (int kt = 0; kt < 4; ++kt)
            o += pr[kt] * (float)sKV[kt * 16 + s][64 + h * 16 + d0 + d];
          sAT[s][h * 16 + d0 + d] = (_Float16)(o * inv);
        }
      }
      __syncthreads();

      // ---- S5: O projection + residual (CLS rows, nt = wave) ----
      {
        int nt = wv;
        v8f acc = {};
#pragma unroll
        for (int kk = 0; kk < 2; ++kk) {
          v16h a = fragA_ld(&sAT[0][kk * 32], LDX, l);
          v16h b = fragB_ld(p.hWo + (size_t)(nt * 16) * D_ + kk * 32, D_, l);
          acc = wmma16(a, b, acc);
        }
        int n = nt * 16 + (int)(l & 15u);
        float bb = p.bo[n];
        int mb = (int)((l >> 4) * 8u);
#pragma unroll
        for (int r = 0; r < 8; ++r) {
          int m = mb + r;
          sY[m][n] = acc[r] + bb + (float)sCLS[m][n];
        }
      }
      __syncthreads();

      // ---- S6: LN1 -> sH  (4 lanes per row, quad shfl reduction) ----
      if (tid < 64) {
        int m = tid >> 2, q = tid & 3;
        int c0 = q * 16;
        float s1 = 0.f, s2 = 0.f;
#pragma unroll
        for (int c = 0; c < 16; ++c) { float v = sY[m][c0 + c]; s1 += v; s2 += v * v; }
        s1 += __shfl_xor(s1, 1, 32); s2 += __shfl_xor(s2, 1, 32);
        s1 += __shfl_xor(s1, 2, 32); s2 += __shfl_xor(s2, 2, 32);
        float mean = s1 * (1.f / 64.f);
        float var = s2 * (1.f / 64.f) - mean * mean;
        float rs = rsqrtf(var + 1e-5f);
#pragma unroll
        for (int c = 0; c < 16; ++c) {
          int cc = c0 + c;
          sH[m][cc] = (_Float16)((sY[m][cc] - mean) * rs * p.ln1w[cc] + p.ln1b[cc]);
        }
      }
      __syncthreads();

      // ---- S7: FF1 (+ReLU), single 16x16 tile (wave 0) ----
      if (wv == 0) {
        v8f acc = {};
#pragma unroll
        for (int kk = 0; kk < 2; ++kk) {
          v16h a = fragA_ld(&sH[0][kk * 32], LDX, l);
          v16h b = fragB_ld(p.hW1 + kk * 32, D_, l);
          acc = wmma16(a, b, acc);
        }
        int n = (int)(l & 15u);
        float bb = p.b1[n];
        int mb = (int)((l >> 4) * 8u);
#pragma unroll
        for (int r = 0; r < 8; ++r) {
          float v = acc[r] + bb;
          sF[mb + r][n] = (_Float16)(v > 0.f ? v : 0.f);
        }
      }
      __syncthreads();

      // ---- S8: FF2 + residual (K=32 padded) ----
      {
        int nt = wv;
        v16h a = fragA_ld(&sF[0][0], LDF, l);
        v16h b = fragB_ld(p.hW2p + (size_t)(nt * 16) * 32, 32, l);
        v8f acc = {};
        acc = wmma16(a, b, acc);
        int n = nt * 16 + (int)(l & 15u);
        float bb = p.b2[n];
        int mb = (int)((l >> 4) * 8u);
#pragma unroll
        for (int r = 0; r < 8; ++r) {
          int m = mb + r;
          sY[m][n] = acc[r] + bb + (float)sH[m][n];
        }
      }
      __syncthreads();

      // ---- S9: LN2 -> tmp (sAT) + mem accumulate (4 lanes per row) ----
      if (tid < 64) {
        int m = tid >> 2, q = tid & 3;
        int c0 = q * 16;
        float s1 = 0.f, s2 = 0.f;
#pragma unroll
        for (int c = 0; c < 16; ++c) { float v = sY[m][c0 + c]; s1 += v; s2 += v * v; }
        s1 += __shfl_xor(s1, 1, 32); s2 += __shfl_xor(s2, 1, 32);
        s1 += __shfl_xor(s1, 2, 32); s2 += __shfl_xor(s2, 2, 32);
        float mean = s1 * (1.f / 64.f);
        float var = s2 * (1.f / 64.f) - mean * mean;
        float rs = rsqrtf(var + 1e-5f);
#pragma unroll
        for (int c = 0; c < 16; ++c) {
          int cc = c0 + c;
          float t = (sY[m][cc] - mean) * rs * p.ln2w[cc] + p.ln2b[cc];
          sMem[st * 16 + m][cc] += t;
          sAT[m][cc] = (_Float16)t;
        }
      }
      __syncthreads();

      // ---- S10: category projection + scatter (edge kernel only, wave 0) ----
      if (!SELF && wv == 0) {
        v8f acc = {};
#pragma unroll
        for (int kk = 0; kk < 2; ++kk) {
          v16h a = fragA_ld(&sAT[0][kk * 32], LDX, l);
          v16h b = fragB_ld(p.hWcP + kk * 32, D_, l);
          acc = wmma16(a, b, acc);
        }
        int j = (int)(l & 15u);
        int mb = (int)((l >> 4) * 8u);
        if (j < MEM_) {
#pragma unroll
          for (int r = 0; r < 8; ++r) {
            int m = mb + r;
            int e = base + st * 16 + m;
            if (e < p.nE) {
              int et = p.etype[e];
              long long pos =
                  (long long)p.start[et] * MEM_ + p.rank[e] + (long long)mi * p.counts[et];
              p.category[pos * MEM_ + j] = acc[r] + p.bc[j];
              if (j == 0) p.label[pos] = mi;
            }
          }
        }
      }
      __syncthreads();
    }  // mi
  }    // st

  // ---- Hypernetwork GEMM: msg = P @ WmR^T, K = 4096, A built on-the-fly ----
  {
    int nt = wv;  // output column tile: o in [16*wv, 16*wv+16)
    v16h sa[4][2];
#pragma unroll
    for (int mt = 0; mt < 4; ++mt)
#pragma unroll
      for (int kk = 0; kk < 2; ++kk)
        sa[mt][kk] = fragA_ld(&sSrc[mt * 16][kk * 32], LDX, l);
    v8f acc[4] = {};
    const int e0 = (int)(l & 15u);
    const _Float16* wbase = p.WmR + (size_t)(nt * 16) * 64;
    for (int j = 0; j < 64; ++j) {
      const _Float16* wj = wbase + (size_t)j * 4096;
      if (j + 1 < 64) __builtin_prefetch(wj + 4096, 0, 1);  // global_prefetch_b8
      v16h b0 = fragB_ld(wj, 64, l);
      v16h b1 = fragB_ld(wj + 32, 64, l);
      _Float16 m0 = (_Float16)sMem[0 * 16 + e0][j];
      _Float16 m1 = (_Float16)sMem[1 * 16 + e0][j];
      _Float16 m2 = (_Float16)sMem[2 * 16 + e0][j];
      _Float16 m3 = (_Float16)sMem[3 * 16 + e0][j];
      acc[0] = wmma16(scale16(sa[0][0], m0), b0, acc[0]);
      acc[0] = wmma16(scale16(sa[0][1], m0), b1, acc[0]);
      acc[1] = wmma16(scale16(sa[1][0], m1), b0, acc[1]);
      acc[1] = wmma16(scale16(sa[1][1], m1), b1, acc[1]);
      acc[2] = wmma16(scale16(sa[2][0], m2), b0, acc[2]);
      acc[2] = wmma16(scale16(sa[2][1], m2), b1, acc[2]);
      acc[3] = wmma16(scale16(sa[3][0], m3), b0, acc[3]);
      acc[3] = wmma16(scale16(sa[3][1], m3), b1, acc[3]);
    }
    int o = nt * 16 + (int)(l & 15u);
    int mb = (int)((l >> 4) * 8u);
#pragma unroll
    for (int mt = 0; mt < 4; ++mt) {
#pragma unroll
      for (int r = 0; r < 8; ++r) {
        int e = base + mt * 16 + mb + r;
        if (e < total) {
          float v = acc[mt][r];
          if (SELF) {
            float c = p.cnt[e];
            float a = c > 0.f ? p.agg[(size_t)e * D_ + o] / c : 0.f;
            p.node_rep[(size_t)e * D_ + o] = v + a + p.h_bias[o];
          } else {
            atomicAdd(&p.agg[(size_t)p.dst_idx[e] * D_ + o], v);
          }
        }
      }
    }
  }
}

// ---------------- host launcher ----------------

extern "C" void kernel_launch(void* const* d_in, const int* in_sizes, int n_in,
                              void* d_out, int out_size, void* d_ws, size_t ws_size,
                              hipStream_t stream) {
  const float* feat    = (const float*)d_in[0];
  const int*   src_idx = (const int*)d_in[1];
  const int*   dst_idx = (const int*)d_in[2];
  const int*   etype   = (const int*)d_in[3];
  const float* Wqkv = (const float*)d_in[4];
  const float* bqkv = (const float*)d_in[5];
  const float* Wo   = (const float*)d_in[6];
  const float* bo   = (const float*)d_in[7];
  const float* W1   = (const float*)d_in[8];
  const float* b1   = (const float*)d_in[9];
  const float* W2   = (const float*)d_in[10];
  const float* b2   = (const float*)d_in[11];
  const float* ln1w = (const float*)d_in[12];
  const float* ln1b = (const float*)d_in[13];
  const float* ln2w = (const float*)d_in[14];
  const float* ln2b = (const float*)d_in[15];
  const float* Wc   = (const float*)d_in[16];
  const float* bc   = (const float*)d_in[17];
  const float* CLS  = (const float*)d_in[18];
  const float* SEP  = (const float*)d_in[19];
  const float* Wm   = (const float*)d_in[20];
  const float* h_bias = (const float*)d_in[21];

  const int N_ = in_sizes[0] / D_;
  const int E_ = in_sizes[1];

  // workspace carve-out (~8.5 MB)
  char* w = (char*)d_ws;
  size_t off = 0;
  auto take = [&](size_t bytes) -> char* {
    char* ptr = w + off;
    off += (bytes + 255) & ~(size_t)255;
    return ptr;
  };
  _Float16* featH = (_Float16*)take((size_t)N_ * D_ * 2);
  _Float16* WmR   = (_Float16*)take((size_t)64 * 4096 * 2);
  _Float16* hWqkv = (_Float16*)take(192 * 64 * 2);
  _Float16* hWo   = (_Float16*)take(64 * 64 * 2);
  _Float16* hW1   = (_Float16*)take(16 * 64 * 2);
  _Float16* hW2p  = (_Float16*)take(64 * 32 * 2);
  _Float16* hWcP  = (_Float16*)take(16 * 64 * 2);
  _Float16* hCLS  = (_Float16*)take(MEM_ * D_ * 2);
  _Float16* hSEP  = (_Float16*)take(D_ * 2);
  float* agg  = (float*)take((size_t)N_ * D_ * 4);
  float* cnt  = (float*)take((size_t)N_ * 4);
  int*   rank = (int*)take((size_t)E_ * 4);
  int*   cnts = (int*)take(NREL_ * 4);
  int*   strt = (int*)take(NREL_ * 4);

  // outputs: node_rep [N,64] f32 | label [MEM*E] i32 | category [MEM*E, MEM] f32
  float* outf     = (float*)d_out;
  float* node_rep = outf;
  int*   label    = (int*)(outf + (size_t)N_ * D_);
  float* category = outf + (size_t)N_ * D_ + (size_t)MEM_ * E_;

  prep_weights<<<512, 256, 0, stream>>>(Wqkv, Wo, W1, W2, Wc, CLS, SEP, Wm, feat,
                                        hWqkv, hWo, hW1, hW2p, hWcP, hCLS, hSEP,
                                        WmR, featH, N_);
  zero_ws<<<1024, 256, 0, stream>>>(agg, cnt, N_);
  rank_kernel<<<1, 256, 0, stream>>>(etype, E_, rank, cnts, strt);
  count_edges<<<(E_ + 255) / 256, 256, 0, stream>>>(dst_idx, cnt, E_);

  GP gp;
  gp.featH = featH; gp.WmR = WmR; gp.hWqkv = hWqkv; gp.hWo = hWo; gp.hW1 = hW1;
  gp.hW2p = hW2p; gp.hWcP = hWcP; gp.hCLS = hCLS; gp.hSEP = hSEP;
  gp.bqkv = bqkv; gp.bo = bo; gp.b1 = b1; gp.b2 = b2;
  gp.ln1w = ln1w; gp.ln1b = ln1b; gp.ln2w = ln2w; gp.ln2b = ln2b;
  gp.bc = bc; gp.h_bias = h_bias;
  gp.src_idx = src_idx; gp.dst_idx = dst_idx; gp.etype = etype;
  gp.rank = rank; gp.counts = cnts; gp.start = strt;
  gp.agg = agg; gp.cnt = cnt;
  gp.node_rep = node_rep; gp.label = label; gp.category = category;
  gp.nE = E_; gp.nN = N_;

  gnn_main<false><<<(E_ + EPW - 1) / EPW, BLK, 0, stream>>>(gp);  // edges -> agg, category
  gnn_main<true><<<(N_ + EPW - 1) / EPW, BLK, 0, stream>>>(gp);   // self-loop + final mean
}
